// RoutedExperts_8830452760817
// MI455X (gfx1250) — matrix operand order repaired and checked by reading, not compiled, not run on hardware
//
#include <hip/hip_runtime.h>
#include <stdint.h>
#include <stddef.h>

// MoE routed-experts forward for MI455X (gfx1250, wave32, WMMA).
// GEMM1: up_gate = x @ Wug^T, h = up * silu(gate)  (h stored bf16 in d_ws)
// GEMM2: out = h @ Wdn                              (f32 output)
// f32 global loads -> native f32->bf16 cvt -> bf16 LDS tiles -> v_wmma_f32_16x16x32_bf16.
// GEMM2 A-tiles (already bf16) are moved with global_load_async_to_lds_b128 (ASYNCcnt).
// K-tile = 64: 16 WMMAs per wave per workgroup barrier, 73.7 KB LDS (CDNA5: 320 KB/WGP).

#define D_MODEL 2048
#define HIDDEN 2048
#define N_EXPERTS 16
#define TOK_PER_E 2048
#define LDT 72  // LDS row stride in halfwords: 64 K-elements + 8 pad (144B, 16B-aligned)

typedef __attribute__((ext_vector_type(16))) __bf16 bf16x16;
typedef __attribute__((ext_vector_type(4)))  __bf16 bf16x4;
typedef __attribute__((ext_vector_type(8)))  float  f32x8;
typedef __attribute__((ext_vector_type(4)))  float  f32x4;
typedef __attribute__((ext_vector_type(4)))  unsigned int   u32x4;
typedef __attribute__((ext_vector_type(4)))  unsigned short u16x4;

static __device__ __forceinline__ u16x4 cvt4(f32x4 v) {
  bf16x4 b;
  b[0] = (__bf16)v[0]; b[1] = (__bf16)v[1];
  b[2] = (__bf16)v[2]; b[3] = (__bf16)v[3];
  return __builtin_bit_cast(u16x4, b);
}
static __device__ __forceinline__ unsigned short cvt1(float v) {
  return __builtin_bit_cast(unsigned short, (__bf16)v);
}

// A fragment (16x32 bf16, ISA 7.12.2): lanes 0-15 rows M hold K {0..7,16..23},
// lanes 16-31 same rows hold K {8..15,24..31}. rowp -> LDS row start (K contiguous).
static __device__ __forceinline__ bf16x16 frag_a(const unsigned short* rowp, int kk, int kh) {
  union { bf16x16 bf; u32x4 q[2]; } u;
  u.q[0] = *(const u32x4*)(rowp + kk + kh * 8);
  u.q[1] = *(const u32x4*)(rowp + kk + 16 + kh * 8);
  return u.bf;
}
// B fragment (32x16 bf16): lane n (0-15) = column N holds K {0..7,16..23}, lanes 16-31 hold K {8..15,24..31}.
static __device__ __forceinline__ bf16x16 frag_b(const unsigned short* rowp, int kk, int kh) {
  union { bf16x16 bf; u32x4 q[2]; } u;
  u.q[0] = *(const u32x4*)(rowp + kk + kh * 8);
  u.q[1] = *(const u32x4*)(rowp + kk + 16 + kh * 8);
  return u.bf;
}

#define WMMA_BF16(a, b, c) \
  __builtin_amdgcn_wmma_f32_16x16x32_bf16(false, (a), false, (b), (short)0, (c), false, false)

// Async copy: 16B from global (saddr base + 32-bit voffset) straight into LDS, no VGPR data.
static __device__ __forceinline__ void async_copy_b128(unsigned lds_off, unsigned goff,
                                                       const void* sbase) {
  asm volatile("global_load_async_to_lds_b128 %0, %1, %2"
               :: "v"(lds_off), "v"(goff), "s"(sbase) : "memory");
}
static __device__ __forceinline__ void wait_async0() {
  asm volatile("s_wait_asynccnt 0x0" ::: "memory");
}

// ---------------- GEMM1: up/gate + SiLU ----------------
// Block tile: 128 tokens x 64 h-cols (64 up cols + 64 gate cols), K-tile 64.
// 8 waves: 4 M-waves x 2 N-waves; wave tile 32x32 (x2 for up/gate).
__global__ __launch_bounds__(256) void moe_up_gate_silu(
    const float* __restrict__ x, const float* __restrict__ wug,
    unsigned short* __restrict__ h) {
  __shared__ unsigned short As[2][128 * LDT];
  __shared__ unsigned short Bs[2][128 * LDT];  // rows 0-63: up weights, 64-127: gate weights

  const int tid = threadIdx.x;
  const int bid = blockIdx.x;
  const int nt = bid & 31;          // 32 n-tiles of 64 h-cols
  const int mt = (bid >> 5) & 15;   // 16 m-tiles of 128 tokens
  const int e  = bid >> 9;          // expert
  const int n0h = nt * 64;

  const float* xe = x + ((size_t)e * TOK_PER_E + (size_t)mt * 128) * D_MODEL;
  const float* we = wug + (size_t)e * (2 * HIDDEN) * D_MODEL;

  const int lane = tid & 31;
  const int wid  = tid >> 5;
  const int mw = wid & 3;
  const int nw = wid >> 2;
  const int rl = lane & 15;
  const int kh = lane >> 4;

  f32x8 accu[2][2], accg[2][2];
#pragma unroll
  for (int i = 0; i < 2; ++i)
#pragma unroll
    for (int j = 0; j < 2; ++j)
#pragma unroll
      for (int v = 0; v < 8; ++v) { accu[i][j][v] = 0.f; accg[i][j][v] = 0.f; }

  auto load_tiles = [&](int buf, int kk0) {
#pragma unroll
    for (int p = 0; p < 8; ++p) {
      int idx = p * 256 + tid;   // 2048 float4 groups cover 128x64
      int row = idx >> 4;
      int cv  = idx & 15;
      f32x4 va = *(const f32x4*)(xe + (size_t)row * D_MODEL + kk0 + cv * 4);
      *(u16x4*)(&As[buf][row * LDT + cv * 4]) = cvt4(va);

      int wrow = (row < 64) ? (n0h + row) : (HIDDEN + n0h + row - 64);
      f32x4 vb = *(const f32x4*)(we + (size_t)wrow * D_MODEL + kk0 + cv * 4);
      *(u16x4*)(&Bs[buf][row * LDT + cv * 4]) = cvt4(vb);
    }
    // Prefetch next-next K-tile on both streams (speculative, no counters).
    if (kk0 + 128 < D_MODEL) {
      int row = tid >> 1, q = (tid & 1) * 32;
      int wrow = (row < 64) ? (n0h + row) : (HIDDEN + n0h + row - 64);
      __builtin_prefetch(xe + (size_t)row * D_MODEL + kk0 + 128 + q, 0, 0);
      __builtin_prefetch(we + (size_t)wrow * D_MODEL + kk0 + 128 + q, 0, 0);
    }
  };

  auto compute = [&](int buf) {
    const unsigned short* A = As[buf];
    const unsigned short* B = Bs[buf];
#pragma unroll
    for (int kk = 0; kk < 64; kk += 32) {
      bf16x16 a0 = frag_a(A + (mw * 32 + rl) * LDT, kk, kh);
      bf16x16 a1 = frag_a(A + (mw * 32 + 16 + rl) * LDT, kk, kh);
#pragma unroll
      for (int s = 0; s < 2; ++s) {
        bf16x16 bu = frag_b(B + (nw * 32 + s * 16 + rl) * LDT, kk, kh);
        bf16x16 bg = frag_b(B + (64 + nw * 32 + s * 16 + rl) * LDT, kk, kh);
        accu[0][s] = WMMA_BF16(a0, bu, accu[0][s]);
        accu[1][s] = WMMA_BF16(a1, bu, accu[1][s]);
        accg[0][s] = WMMA_BF16(a0, bg, accg[0][s]);
        accg[1][s] = WMMA_BF16(a1, bg, accg[1][s]);
      }
    }
  };

  load_tiles(0, 0);
  __syncthreads();
  int buf = 0;
  for (int kt = 0; kt < D_MODEL / 64; ++kt) {
    if (kt + 1 < D_MODEL / 64) load_tiles(buf ^ 1, (kt + 1) * 64);
    compute(buf);
    __syncthreads();
    buf ^= 1;
  }

  // Epilogue: h = up * silu(gate), store bf16. C layout: VGPR v -> M = v + 8*kh, N = rl.
#pragma unroll
  for (int ms = 0; ms < 2; ++ms) {
#pragma unroll
    for (int s = 0; s < 2; ++s) {
      size_t tok = (size_t)e * TOK_PER_E + mt * 128 + mw * 32 + ms * 16 + kh * 8;
      int col = n0h + nw * 32 + s * 16 + rl;
#pragma unroll
      for (int v = 0; v < 8; ++v) {
        float up = accu[ms][s][v];
        float g  = accg[ms][s][v];
        float hv = up * (g / (1.0f + __expf(-g)));
        h[(tok + v) * HIDDEN + col] = cvt1(hv);
      }
    }
  }
}

// ---------------- GEMM2: down projection ----------------
// Block tile: 128 tokens x 128 d-cols, K-tile 64. 8 waves: 4 M x 2 N; wave tile 32x64.
// A-tile (bf16 h) streamed via global_load_async_to_lds_b128.
__global__ __launch_bounds__(256) void moe_down(
    const unsigned short* __restrict__ h, const float* __restrict__ wd,
    float* __restrict__ out) {
  __shared__ unsigned short As[2][128 * LDT];
  __shared__ unsigned short Bt[2][128 * LDT];  // B transposed: row = d-col, K contiguous

  const int tid = threadIdx.x;
  const int bid = blockIdx.x;
  const int nt = bid & 15;          // 16 n-tiles of 128
  const int mt = (bid >> 4) & 15;   // 16 m-tiles of 128
  const int e  = bid >> 8;
  const int n0 = nt * 128;

  const unsigned short* he = h + ((size_t)e * TOK_PER_E + (size_t)mt * 128) * HIDDEN;
  const float* wde = wd + (size_t)e * HIDDEN * D_MODEL;

  const int lane = tid & 31;
  const int wid  = tid >> 5;
  const int mw = wid & 3;
  const int nw = wid >> 2;
  const int n0w = nw * 64;
  const int rl = lane & 15;
  const int kh = lane >> 4;

  f32x8 acc[2][4];
#pragma unroll
  for (int i = 0; i < 2; ++i)
#pragma unroll
    for (int j = 0; j < 4; ++j)
#pragma unroll
      for (int v = 0; v < 8; ++v) acc[i][j][v] = 0.f;

  auto load_tiles = [&](int buf, int kk0) {
    // A: 128x64 bf16, async DMA global -> LDS (16B per lane, 4 issues/thread)
#pragma unroll
    for (int p = 0; p < 4; ++p) {
      int idx = p * 256 + tid;   // 1024 groups of 8 halfwords
      int row = idx >> 3;
      int g8  = idx & 7;
      unsigned lds_off = (unsigned)(uintptr_t)(&As[buf][row * LDT + g8 * 8]);
      unsigned goff = (unsigned)((row * HIDDEN + kk0 + g8 * 8) * 2);  // bytes
      async_copy_b128(lds_off, goff, (const void*)he);
    }
    // B: 64(k) x 128(n) f32, cvt + transpose into Bt[n][k] bf16
#pragma unroll
    for (int p = 0; p < 8; ++p) {
      int idx = p * 256 + tid;   // 2048 float4 groups
      int kr = idx >> 5;         // k row 0..63
      int gc = idx & 31;         // group of 4 n-cols
      f32x4 v = *(const f32x4*)(wde + (size_t)(kk0 + kr) * D_MODEL + n0 + gc * 4);
#pragma unroll
      for (int i = 0; i < 4; ++i)
        Bt[buf][(gc * 4 + i) * LDT + kr] = cvt1(v[i]);
    }
    // Prefetch weight stream one tile beyond the next.
    if (kk0 + 128 < HIDDEN) {
      int kr = tid >> 2, q = (tid & 3) * 32;
      __builtin_prefetch(wde + (size_t)(kk0 + 128 + kr) * D_MODEL + n0 + q, 0, 0);
    }
  };

  auto compute = [&](int buf) {
    const unsigned short* A = As[buf];
    const unsigned short* B = Bt[buf];
#pragma unroll
    for (int kk = 0; kk < 64; kk += 32) {
      bf16x16 a0 = frag_a(A + (mw * 32 + rl) * LDT, kk, kh);
      bf16x16 a1 = frag_a(A + (mw * 32 + 16 + rl) * LDT, kk, kh);
#pragma unroll
      for (int s = 0; s < 4; ++s) {
        bf16x16 b = frag_b(B + (n0w + s * 16 + rl) * LDT, kk, kh);
        acc[0][s] = WMMA_BF16(a0, b, acc[0][s]);
        acc[1][s] = WMMA_BF16(a1, b, acc[1][s]);
      }
    }
  };

  load_tiles(0, 0);
  wait_async0();
  __syncthreads();
  int buf = 0;
  for (int kt = 0; kt < HIDDEN / 64; ++kt) {
    if (kt + 1 < HIDDEN / 64) load_tiles(buf ^ 1, (kt + 1) * 64);
    compute(buf);
    wait_async0();   // async LDS writes for next buffer complete before the barrier
    __syncthreads();
    buf ^= 1;
  }

#pragma unroll
  for (int ms = 0; ms < 2; ++ms) {
#pragma unroll
    for (int s = 0; s < 4; ++s) {
      size_t tok = (size_t)e * TOK_PER_E + mt * 128 + mw * 32 + ms * 16 + kh * 8;
      int col = n0 + n0w + s * 16 + rl;
#pragma unroll
      for (int v = 0; v < 8; ++v)
        out[(tok + v) * D_MODEL + col] = acc[ms][s][v];
    }
  }
}

extern "C" void kernel_launch(void* const* d_in, const int* in_sizes, int n_in,
                              void* d_out, int out_size, void* d_ws, size_t ws_size,
                              hipStream_t stream) {
  (void)in_sizes; (void)n_in; (void)out_size; (void)ws_size;
  const float* x   = (const float*)d_in[0];
  const float* wug = (const float*)d_in[1];
  const float* wd  = (const float*)d_in[2];
  // d_in[3] = batch_size_per_expert: equal split (2048/expert), statically known.
  float* out = (float*)d_out;
  unsigned short* h = (unsigned short*)d_ws;  // 32768 x 2048 bf16 = 128 MB scratch

  moe_up_gate_silu<<<dim3(N_EXPERTS * 16 * 32), dim3(256), 0, stream>>>(x, wug, h);
  moe_down<<<dim3(N_EXPERTS * 16 * 16), dim3(256), 0, stream>>>(h, wd, out);
}
